// CGT_21354577396059
// MI455X (gfx1250) — compile-verified, matrix operand order checked by reading, not run on hardware
//
#include <hip/hip_runtime.h>
#include <hip/hip_bf16.h>
#include <stdint.h>

typedef __bf16 bf16_t;
typedef __attribute__((ext_vector_type(8)))  __bf16 v8bf;
typedef __attribute__((ext_vector_type(16))) __bf16 v16bf;
typedef __attribute__((ext_vector_type(8)))  float  v8f;

#define NN   16384
#define CC   128
#define NE   262144
#define NG   64
#define NPG  256
#define HD   32
#define LYR  3
#define BM   128
#define BN   128
#define KB   32
#define BN_INV 0.9999950000374998f   // 1/sqrt(1+1e-5)

// Switch for the CDNA5 async global->LDS data mover path.
#define USE_ASYNC_LDS 1

// ---------------------------------------------------------------------------
// CDNA5 async copy: GLOBAL_LOAD_ASYNC_TO_LDS_B128 (ASYNCcnt-tracked, writes
// LDS directly, no VGPR round-trip).  VDST operand = LDS byte offset (low 32
// bits of the generic LDS pointer per the aperture mapping), VADDR = 64-bit
// global address.
// ---------------------------------------------------------------------------
static __device__ inline void cp16_g2l(bf16_t* l, const bf16_t* g) {
#if USE_ASYNC_LDS
  asm volatile("global_load_async_to_lds_b128 %0, %1, off"
               :: "v"((uint32_t)(uintptr_t)l), "v"((uint64_t)(uintptr_t)g)
               : "memory");
#else
  *(v8bf*)l = *(const v8bf*)g;
#endif
}

static __device__ inline void async_wait() {
#if USE_ASYNC_LDS
  asm volatile("s_wait_asynccnt 0x0" ::: "memory");
#endif
}

// ---------------------------------------------------------------------------
// WMMA helpers (CDNA5 wave32, 16x16x32 bf16 -> f32)
// ---------------------------------------------------------------------------
static __device__ inline v16bf ldfrag(const bf16_t* base, int stride, int lane) {
  // A-style fragment: lane m = lane&15 holds row m; K pairs split lo/hi half-wave.
  int m  = lane & 15;
  int kb = (lane & 16) ? 8 : 0;
  const bf16_t* p = base + m * stride + kb;
  union { v16bf v; v8bf part[2]; } u;
  u.part[0] = *(const v8bf*)(p);
  u.part[1] = *(const v8bf*)(p + 16);
  return u.v;
}

static __device__ inline v8f wmma_bf16(v16bf a, v16bf b, v8f c) {
  return __builtin_amdgcn_wmma_f32_16x16x32_bf16(false, a, false, b, (short)0, c,
                                                 false, false);
}

// ---------------------------------------------------------------------------
// Weight prep: fp32 [K][N] (trans=1) or [N][K] (trans=0) -> bf16 [N][K]
// ---------------------------------------------------------------------------
__global__ void k_prep(const float* __restrict__ src, bf16_t* __restrict__ dst,
                       int K, int Nn, int trans) {
  int idx = blockIdx.x * 256 + threadIdx.x;
  if (idx >= K * Nn) return;
  int n = idx / K, k = idx - n * K;
  float v = trans ? src[(size_t)k * Nn + n] : src[(size_t)n * K + k];
  dst[idx] = (bf16_t)v;
}

// ---------------------------------------------------------------------------
// Embedding: h = x @ We + be + pe @ Wp + bp   (tiny K: VALU)
// ---------------------------------------------------------------------------
__global__ void k_embed(const float* __restrict__ x, const float* __restrict__ pe,
                        const float* __restrict__ We, const float* __restrict__ be,
                        const float* __restrict__ Wp, const float* __restrict__ bp,
                        bf16_t* __restrict__ h) {
  int idx = blockIdx.x * 256 + threadIdx.x;       // NN*CC threads
  int n = idx >> 7, c = idx & 127;
  float acc = be[c] + bp[c];
  const float* xr = x + (size_t)n * 32;
  const float* pr = pe + (size_t)n * 20;
#pragma unroll 8
  for (int k = 0; k < 32; ++k) acc += xr[k] * We[k * CC + c];
#pragma unroll 4
  for (int k = 0; k < 20; ++k) acc += pr[k] * Wp[k * CC + c];
  h[idx] = (bf16_t)acc;
}

// ---------------------------------------------------------------------------
// GIN edge aggregation: agg[dst] += h[src]   (fp32 atomics)
// ---------------------------------------------------------------------------
__global__ void k_scatter(const bf16_t* __restrict__ h, const int* __restrict__ ei,
                          float* __restrict__ agg) {
  long t = (long)blockIdx.x * 256 + threadIdx.x;  // NE*32 threads, 4 ch each
  int e = (int)(t >> 5), q = ((int)t & 31) * 4;
  int src = ei[e], dst = ei[NE + e];
  const bf16_t* hp = h + (size_t)src * CC + q;
  float* ap = agg + (size_t)dst * CC + q;
#pragma unroll
  for (int i = 0; i < 4; ++i) atomicAdd(ap + i, (float)hp[i]);
}

// ---------------------------------------------------------------------------
// Elementwise fusions (bf16 activations, per-channel BN affine)
// ---------------------------------------------------------------------------
__global__ void k_add_hagg(const bf16_t* __restrict__ h, const float* __restrict__ agg,
                           bf16_t* __restrict__ o) {
  int i = blockIdx.x * 256 + threadIdx.x;
  o[i] = (bf16_t)((float)h[i] + agg[i]);
}

// o = (x + y) * bn_w*inv + bn_b
__global__ void k_bnadd(const bf16_t* __restrict__ x, const bf16_t* __restrict__ y,
                        const float* __restrict__ w, const float* __restrict__ b,
                        bf16_t* __restrict__ o) {
  int i = blockIdx.x * 256 + threadIdx.x;
  int c = i & 127;
  o[i] = (bf16_t)(((float)x[i] + (float)y[i]) * (w[c] * BN_INV) + b[c]);
}

// o = h1 + ((a + h) * bn_w*inv + bn_b)
__global__ void k_fuse_out(const bf16_t* __restrict__ h1, const bf16_t* __restrict__ a,
                           const bf16_t* __restrict__ h, const float* __restrict__ w,
                           const float* __restrict__ b, bf16_t* __restrict__ o) {
  int i = blockIdx.x * 256 + threadIdx.x;
  int c = i & 127;
  o[i] = (bf16_t)((float)h1[i] + (((float)a[i] + (float)h[i]) * (w[c] * BN_INV) + b[c]));
}

// h = h + ((out + m2) * bn_w*inv + bn_b)
__global__ void k_final(bf16_t* __restrict__ h, const bf16_t* __restrict__ ob,
                        const bf16_t* __restrict__ m2, const float* __restrict__ w,
                        const float* __restrict__ b) {
  int i = blockIdx.x * 256 + threadIdx.x;
  int c = i & 127;
  h[i] = (bf16_t)((float)h[i] + (((float)ob[i] + (float)m2[i]) * (w[c] * BN_INV) + b[c]));
}

// ---------------------------------------------------------------------------
// WMMA GEMM: C[M][N] = act(A[M][K]bf16 @ Wt[N][K]^T + bias), bf16 out.
// 256 threads = 8 waves; block tile 128x128; wave tile 32x64 (2x4 WMMA tiles).
// Double-buffered LDS staging via async global->LDS copies: the next tile's
// copies are in flight (ASYNCcnt) while WMMA consumes the current tile.
// ---------------------------------------------------------------------------
__global__ __launch_bounds__(256) void k_gemm(const bf16_t* __restrict__ A,
                                              const bf16_t* __restrict__ Wt,
                                              const float* __restrict__ bias,
                                              bf16_t* __restrict__ Cout,
                                              int K, int Nout, int relu) {
  __shared__ __align__(16) bf16_t As[2][BM * KB];
  __shared__ __align__(16) bf16_t Bs[2][BN * KB];
  int tid = threadIdx.x, lane = tid & 31, w = tid >> 5;
  int rowA0 = blockIdx.x * BM;
  int colB0 = blockIdx.y * BN;
  int wm = (w & 3) * 32, wn = (w >> 2) * 64;
  v8f acc[2][4] = {};
  int nk = K / KB;

  // prologue fill buffer 0
#pragma unroll
  for (int i = 0; i < 2; ++i) {
    int cch = tid + i * 256;                  // 512 x 16B chunks
    int r = cch >> 2, col = (cch & 3) * 8;
    cp16_g2l(As[0] + r * KB + col, A + (size_t)(rowA0 + r) * K + col);
    cp16_g2l(Bs[0] + r * KB + col, Wt + (size_t)(colB0 + r) * K + col);
  }
  async_wait();
  __syncthreads();

  for (int kk = 0; kk < nk; ++kk) {
    int cur = kk & 1;
    if (kk + 1 < nk) {
      int kN = (kk + 1) * KB;
#pragma unroll
      for (int i = 0; i < 2; ++i) {
        int cch = tid + i * 256;
        int r = cch >> 2, col = (cch & 3) * 8;
        cp16_g2l(As[cur ^ 1] + r * KB + col,
                 A + (size_t)(rowA0 + r) * K + kN + col);
        cp16_g2l(Bs[cur ^ 1] + r * KB + col,
                 Wt + (size_t)(colB0 + r) * K + kN + col);
      }
    }
    v16bf af[2], bfg[4];
#pragma unroll
    for (int r = 0; r < 2; ++r) af[r] = ldfrag(As[cur] + (wm + r * 16) * KB, KB, lane);
#pragma unroll
    for (int c = 0; c < 4; ++c) bfg[c] = ldfrag(Bs[cur] + (wn + c * 16) * KB, KB, lane);
#pragma unroll
    for (int r = 0; r < 2; ++r)
#pragma unroll
      for (int c = 0; c < 4; ++c)
        acc[r][c] = wmma_bf16(af[r], bfg[c], acc[r][c]);
    async_wait();          // this wave's async copies into the next buffer landed
    __syncthreads();
  }

  // epilogue: C/D layout -> global (lane holds col n=lane&15; vgpr j row = j (+8 hi half))
#pragma unroll
  for (int r = 0; r < 2; ++r) {
    int mbase = rowA0 + wm + r * 16 + ((lane & 16) >> 1);
#pragma unroll
    for (int c = 0; c < 4; ++c) {
      int col = colB0 + wn + c * 16 + (lane & 15);
      float bv = bias ? bias[col] : 0.0f;
#pragma unroll
      for (int j = 0; j < 8; ++j) {
        float v = acc[r][c][j] + bv;
        if (relu) v = fmaxf(v, 0.0f);
        Cout[(size_t)(mbase + j) * Nout + col] = (bf16_t)v;
      }
    }
  }
}

// ---------------------------------------------------------------------------
// Fused flash attention per (graph, head): S=256, hd=32, 4 waves x 64 rows.
// qkv: [NN][384] bf16 (q|k|v). out: [NN][128] bf16 (per-head slice).
// ---------------------------------------------------------------------------
__global__ __launch_bounds__(128) void k_attn(const bf16_t* __restrict__ qkv,
                                              bf16_t* __restrict__ out) {
  __shared__ __align__(16) bf16_t Ks[NPG * HD];      // [256][32]
  __shared__ __align__(16) bf16_t Vt[HD * NPG];      // [32][256] (transposed)
  __shared__ __align__(16) bf16_t Pscr[4][16 * 32];  // per-wave P relayout scratch
  int g = blockIdx.x >> 2, hh = blockIdx.x & 3;
  int tid = threadIdx.x, lane = tid & 31, w = tid >> 5;
  size_t base = (size_t)g * NPG;

  // stage K (row-major, async direct-to-LDS) and V (transposed) head slices
  for (int i = tid; i < NPG * HD / 8; i += 128) {
    int r = i >> 2, col = (i & 3) * 8;
    cp16_g2l(Ks + r * HD + col,
             qkv + (base + r) * 384 + 128 + hh * HD + col);
  }
  for (int i = tid; i < NPG * HD; i += 128) {
    int r = i >> 5, c = i & 31;
    Vt[c * NPG + r] = qkv[(base + r) * 384 + 256 + hh * HD + c];
  }
  async_wait();
  __syncthreads();

  int mrow = lane & 15;
  int kb = (lane & 16) ? 8 : 0;

  for (int qt = 0; qt < 4; ++qt) {
    int q0 = w * 64 + qt * 16;
    // Q fragment straight from global (A layout)
    const bf16_t* qp = qkv + (base + q0 + mrow) * 384 + hh * HD + kb;
    v16bf Qf;
    { union { v16bf v; v8bf p[2]; } u;
      u.p[0] = *(const v8bf*)qp; u.p[1] = *(const v8bf*)(qp + 16); Qf = u.v; }

    v8f O0 = {}, O1 = {};
    float mi[8], li[8];
#pragma unroll
    for (int j = 0; j < 8; ++j) { mi[j] = -1e30f; li[j] = 0.0f; }

    for (int kt = 0; kt < 8; ++kt) {          // 32 keys per step
      int n0 = kt * 32;
      v16bf Bk0 = ldfrag(Ks + n0 * HD, HD, lane);
      v16bf Bk1 = ldfrag(Ks + (n0 + 16) * HD, HD, lane);
      v8f Z = {};
      v8f S0 = wmma_bf16(Qf, Bk0, Z);
      v8f S1 = wmma_bf16(Qf, Bk1, Z);

      bf16_t* pw = Pscr[w];
#pragma unroll
      for (int j = 0; j < 8; ++j) {
        float s0 = S0[j] * 0.17677669529663687f;
        float s1 = S1[j] * 0.17677669529663687f;
        float v = fmaxf(s0, s1);
        for (int o = 8; o >= 1; o >>= 1) v = fmaxf(v, __shfl_xor(v, o, 32));
        float mn = fmaxf(mi[j], v);
        float corr = __expf(mi[j] - mn);
        float p0 = __expf(s0 - mn), p1 = __expf(s1 - mn);
        float rs = p0 + p1;
        for (int o = 8; o >= 1; o >>= 1) rs += __shfl_xor(rs, o, 32);
        li[j] = li[j] * corr + rs;
        mi[j] = mn;
        O0[j] *= corr; O1[j] *= corr;
        int rr = j + ((lane & 16) >> 1);
        pw[rr * 32 + (lane & 15)]      = (bf16_t)p0;
        pw[rr * 32 + 16 + (lane & 15)] = (bf16_t)p1;
      }
      // P (A layout, 16x32) @ V (32x32) accumulated into O
      v16bf Pf  = ldfrag(pw, 32, lane);
      v16bf Bv0 = ldfrag(Vt + n0, NPG, lane);            // out cols 0..15
      v16bf Bv1 = ldfrag(Vt + 16 * NPG + n0, NPG, lane); // out cols 16..31
      O0 = wmma_bf16(Pf, Bv0, O0);
      O1 = wmma_bf16(Pf, Bv1, O1);
    }

    // normalize and write per-head output slice
#pragma unroll
    for (int j = 0; j < 8; ++j) {
      float inv = 1.0f / li[j];
      size_t node = base + q0 + j + ((lane & 16) >> 1);
      out[node * CC + hh * HD + (lane & 15)]      = (bf16_t)(O0[j] * inv);
      out[node * CC + hh * HD + 16 + (lane & 15)] = (bf16_t)(O1[j] * inv);
    }
  }
}

// ---------------------------------------------------------------------------
// Pool: per-graph sum over NPG nodes (fp32 out)
// ---------------------------------------------------------------------------
__global__ void k_pool(const bf16_t* __restrict__ h, float* __restrict__ dst) {
  int g = blockIdx.x, c = threadIdx.x;
  const bf16_t* p = h + (size_t)g * NPG * CC + c;
  float s = 0.0f;
  for (int i = 0; i < NPG; ++i) s += (float)p[(size_t)i * CC];
  dst[(size_t)g * CC + c] = s;
}

// ---------------------------------------------------------------------------
// Host orchestration
// ---------------------------------------------------------------------------
extern "C" void kernel_launch(void* const* d_in, const int* in_sizes, int n_in,
                              void* d_out, int out_size, void* d_ws, size_t ws_size,
                              hipStream_t stream) {
  const float* x   = (const float*)d_in[0];
  const float* x1  = (const float*)d_in[1];
  const float* x2  = (const float*)d_in[2];
  const float* pe  = (const float*)d_in[3];
  const int* ei[3] = { (const int*)d_in[4], (const int*)d_in[5], (const int*)d_in[6] };
  const float* We  = (const float*)d_in[9];
  const float* be  = (const float*)d_in[10];
  const float* Wp  = (const float*)d_in[11];
  const float* bp  = (const float*)d_in[12];
  const float* gw1 = (const float*)d_in[13];
  const float* gb1 = (const float*)d_in[14];
  const float* gw2 = (const float*)d_in[15];
  const float* gb2 = (const float*)d_in[16];
  const float* aiw = (const float*)d_in[17];
  const float* aib = (const float*)d_in[18];
  const float* aow = (const float*)d_in[19];
  const float* aob = (const float*)d_in[20];
  const float* m1w = (const float*)d_in[21];
  const float* m1b = (const float*)d_in[22];
  const float* m2w = (const float*)d_in[23];
  const float* m2b = (const float*)d_in[24];
  const float* bnw = (const float*)d_in[25];
  const float* bnb = (const float*)d_in[26];
  float* out = (float*)d_out;

  size_t off = 0;
  auto carve = [&](size_t bytes) -> void* {
    void* p = (void*)((char*)d_ws + off);
    off += (bytes + 255) & ~(size_t)255;
    return p;
  };
  const size_t NC = (size_t)NN * CC;
  bf16_t* wg1t = (bf16_t*)carve((size_t)LYR * 128 * 128 * 2);
  bf16_t* wg2t = (bf16_t*)carve((size_t)LYR * 128 * 128 * 2);
  bf16_t* wait_ = (bf16_t*)carve((size_t)LYR * 384 * 128 * 2);
  bf16_t* waot = (bf16_t*)carve((size_t)LYR * 128 * 128 * 2);
  bf16_t* wm1t = (bf16_t*)carve((size_t)LYR * 256 * 128 * 2);
  bf16_t* wm2t = (bf16_t*)carve((size_t)LYR * 128 * 256 * 2);
  bf16_t* hbuf[3];
  for (int s = 0; s < 3; ++s) hbuf[s] = (bf16_t*)carve(NC * 2);
  float*  agg  = (float*)carve(NC * 4);
  bf16_t* gin  = (bf16_t*)carve(NC * 2);
  bf16_t* t1   = (bf16_t*)carve(NC * 2);
  bf16_t* t2   = (bf16_t*)carve(NC * 2);
  bf16_t* h1   = (bf16_t*)carve(NC * 2);
  bf16_t* qkvb = (bf16_t*)carve((size_t)NN * 384 * 2);
  bf16_t* aout = (bf16_t*)carve(NC * 2);
  bf16_t* abuf = (bf16_t*)carve(NC * 2);
  bf16_t* outb = (bf16_t*)carve(NC * 2);
  bf16_t* mm1  = (bf16_t*)carve((size_t)NN * 256 * 2);
  bf16_t* mm2  = (bf16_t*)carve(NC * 2);

  // ---- weight prep (bf16, [N][K]) ----
  for (int l = 0; l < LYR; ++l) {
    k_prep<<<(128 * 128 + 255) / 256, 256, 0, stream>>>(gw1 + (size_t)l * 128 * 128,
        wg1t + (size_t)l * 128 * 128, 128, 128, 1);
    k_prep<<<(128 * 128 + 255) / 256, 256, 0, stream>>>(gw2 + (size_t)l * 128 * 128,
        wg2t + (size_t)l * 128 * 128, 128, 128, 1);
    k_prep<<<(384 * 128 + 255) / 256, 256, 0, stream>>>(aiw + (size_t)l * 384 * 128,
        wait_ + (size_t)l * 384 * 128, 128, 384, 0);
    k_prep<<<(128 * 128 + 255) / 256, 256, 0, stream>>>(aow + (size_t)l * 128 * 128,
        waot + (size_t)l * 128 * 128, 128, 128, 0);
    k_prep<<<(256 * 128 + 255) / 256, 256, 0, stream>>>(m1w + (size_t)l * 128 * 256,
        wm1t + (size_t)l * 256 * 128, 128, 256, 1);
    k_prep<<<(256 * 128 + 255) / 256, 256, 0, stream>>>(m2w + (size_t)l * 256 * 128,
        wm2t + (size_t)l * 128 * 256, 256, 128, 1);
  }

  // ---- embeddings ----
  const int gEW = (int)(NC / 256);
  k_embed<<<gEW, 256, 0, stream>>>(x,  pe, We, be, Wp, bp, hbuf[0]);
  k_embed<<<gEW, 256, 0, stream>>>(x1, pe, We, be, Wp, bp, hbuf[1]);
  k_embed<<<gEW, 256, 0, stream>>>(x2, pe, We, be, Wp, bp, hbuf[2]);

  const int gE = (int)(NC / 256);
  dim3 blk(256);
  for (int l = 0; l < LYR; ++l) {
    const float* bnw_l = bnw + (size_t)l * 3 * CC;
    const float* bnb_l = bnb + (size_t)l * 3 * CC;
    for (int s = 0; s < 3; ++s) {
      bf16_t* h = hbuf[s];
      // GIN aggregation
      hipMemsetAsync(agg, 0, NC * 4, stream);
      k_scatter<<<(int)((size_t)NE * 32 / 256), blk, 0, stream>>>(h, ei[s], agg);
      k_add_hagg<<<gE, blk, 0, stream>>>(h, agg, gin);
      // GIN MLP
      k_gemm<<<dim3(NN / BM, 1), blk, 0, stream>>>(gin, wg1t + (size_t)l * 128 * 128,
          gb1 + (size_t)l * CC, t1, 128, 128, 1);
      k_gemm<<<dim3(NN / BM, 1), blk, 0, stream>>>(t1, wg2t + (size_t)l * 128 * 128,
          gb2 + (size_t)l * CC, t2, 128, 128, 0);
      k_bnadd<<<gE, blk, 0, stream>>>(t2, h, bnw_l, bnb_l, h1);
      // attention branch
      k_gemm<<<dim3(NN / BM, 3), blk, 0, stream>>>(h, wait_ + (size_t)l * 384 * 128,
          aib + (size_t)l * 384, qkvb, 128, 384, 0);
      k_attn<<<NG * 4, 128, 0, stream>>>(qkvb, aout);
      k_gemm<<<dim3(NN / BM, 1), blk, 0, stream>>>(aout, waot + (size_t)l * 128 * 128,
          aob + (size_t)l * CC, abuf, 128, 128, 0);
      k_fuse_out<<<gE, blk, 0, stream>>>(h1, abuf, h, bnw_l + CC, bnb_l + CC, outb);
      // MLP
      k_gemm<<<dim3(NN / BM, 2), blk, 0, stream>>>(outb, wm1t + (size_t)l * 256 * 128,
          m1b + (size_t)l * 2 * CC, mm1, 128, 256, 1);
      k_gemm<<<dim3(NN / BM, 1), blk, 0, stream>>>(mm1, wm2t + (size_t)l * 128 * 256,
          m2b + (size_t)l * CC, mm2, 256, 128, 0);
      k_final<<<gE, blk, 0, stream>>>(h, outb, mm2, bnw_l + 2 * CC, bnb_l + 2 * CC);
    }
  }

  // ---- pooling (3 streams concatenated) ----
  for (int s = 0; s < 3; ++s)
    k_pool<<<NG, CC, 0, stream>>>(hbuf[s], out + (size_t)s * NG * CC);
}
